// RNNClassifier_69458211111685
// MI455X (gfx1250) — matrix operand order repaired and checked by reading, not compiled, not run on hardware
//
#include <hip/hip_runtime.h>
#include <hip/hip_bf16.h>

// B=64, T=2048, V=50000, E=256, H=256, O=16
#define BB 64
#define TT 2048
#define EE 256
#define HH 256
#define OO 16

typedef __attribute__((ext_vector_type(16))) __bf16 v16bf;
typedef __attribute__((ext_vector_type(8)))  __bf16 v8bf;
typedef __attribute__((ext_vector_type(8)))  float  v8f;
typedef __attribute__((ext_vector_type(4)))  float  v4f;
typedef __attribute__((ext_vector_type(4)))  unsigned int v4u;
typedef __attribute__((ext_vector_type(8)))  int v8i;
typedef __attribute__((ext_vector_type(4)))  int v4i;

#if defined(__has_builtin)
#if __has_builtin(__builtin_amdgcn_tensor_load_to_lds)
#define HAVE_TDM 1
#endif
#endif
#ifndef HAVE_TDM
#define HAVE_TDM 0
#endif

// ---------------------------------------------------------------------------
// WMMA helpers (CDNA5 v_wmma_f32_16x16x32_bf16, wave32)
// ---------------------------------------------------------------------------
__device__ __forceinline__ v8f wmma_bf16(v16bf a, v16bf b, v8f c) {
  // (neg_a, A, neg_b, B, c_mod, C, reuse_a, reuse_b)
  return __builtin_amdgcn_wmma_f32_16x16x32_bf16(false, a, false, b,
                                                 (short)0, c, false, false);
}

// A fragment (16x32, 16-bit): lane L holds row M=L%16.
// lanes 0-15: K = kstep*32 + {0..7, 16..23}; lanes 16-31: +8 on both halves.
__device__ __forceinline__ v16bf load_frag_a(const __bf16* row, int kstep, int lane) {
  const int kb = kstep * 32 + ((lane & 16) ? 8 : 0);
  v8bf lo = *(const v8bf*)(row + kb);
  v8bf hi = *(const v8bf*)(row + kb + 16);
  v16bf a;
#pragma unroll
  for (int j = 0; j < 8; ++j) { a[j] = lo[j]; a[j + 8] = hi[j]; }
  return a;
}

// B fragment (32x16, 16-bit): lane L holds col N=L%16.
// lanes 0-15: K = kstep*32 + {0..7, 16..23}; lanes 16-31: +8 on both halves.
// B(k,n) = W[n][k] (computing X @ W.T), so `wrow` = row n of row-major W.
__device__ __forceinline__ v16bf load_frag_b(const __bf16* wrow, int kstep, int lane) {
  const int kb = kstep * 32 + ((lane & 16) ? 8 : 0);
  v8bf lo = *(const v8bf*)(wrow + kb);
  v8bf hi = *(const v8bf*)(wrow + kb + 16);
  v16bf b;
#pragma unroll
  for (int j = 0; j < 8; ++j) { b[j] = lo[j]; b[j + 8] = hi[j]; }
  return b;
}

#if HAVE_TDM
// ---------------------------------------------------------------------------
// Tensor Data Mover: load a [16 x 256] fp32 tile of xh (row stride TT*HH
// elements) for timestep t into LDS at lds_dst. D# per CDNA5 ISA ch.8.
// ---------------------------------------------------------------------------
__device__ __forceinline__ void tdm_load_xh_tile(const float* base, int t,
                                                 float* lds_dst) {
  const unsigned long long ga =
      (unsigned long long)(__UINTPTR_TYPE__)(base + (size_t)t * HH);
  const unsigned int lds_off = (unsigned int)(__UINTPTR_TYPE__)lds_dst;

  v4u g0;
  g0[0] = 1u;                                          // count=1 (valid, user)
  g0[1] = lds_off;                                     // lds_addr (bytes)
  g0[2] = (unsigned int)(ga & 0xFFFFFFFFu);            // global_addr[31:0]
  g0[3] = (unsigned int)((ga >> 32) & 0x1FFFFFFu)      // global_addr[56:32]
          | (2u << 30);                                // type=2 ("image")

  v8i g1;
  g1[0] = 0x00020000;                                  // data_size=2 (4 bytes)
  g1[1] = (int)((HH & 0xFFFF) << 16);                  // tensor_dim0[15:0]
  g1[2] = (int)(((HH >> 16) & 0xFFFF) |                // tensor_dim0[31:16]
                ((16 & 0xFFFF) << 16));                // tensor_dim1[15:0]=16
  g1[3] = (int)(((16 >> 16) & 0xFFFF) |                // tensor_dim1[31:16]
                ((HH & 0xFFFF) << 16));                // tile_dim0=256
  g1[4] = 16;                                          // tile_dim1=16, tile_dim2=0
  g1[5] = (int)(TT * HH);                              // tensor_dim0_stride lo
  g1[6] = 0;                                           // stride hi | dim1_stride lo
  g1[7] = 0;

  v4i g2 = {0, 0, 0, 0};
  v4i g3 = {0, 0, 0, 0};
#if defined(__clang_major__) && (__clang_major__ >= 23)
  v8i g4 = {0, 0, 0, 0, 0, 0, 0, 0};
  __builtin_amdgcn_tensor_load_to_lds(g0, g1, g2, g3, g4, 0);
#else
  __builtin_amdgcn_tensor_load_to_lds(g0, g1, g2, g3, 0);
#endif
}
#endif  // HAVE_TDM

// ---------------------------------------------------------------------------
// k0: fp32 -> bf16 weight conversion
// ---------------------------------------------------------------------------
__global__ void k_f32_to_bf16(const float* __restrict__ src,
                              __bf16* __restrict__ dst, int n) {
  int i = blockIdx.x * blockDim.x + threadIdx.x;
  if (i < n) dst[i] = (__bf16)src[i];
}

// ---------------------------------------------------------------------------
// k1: fused embedding gather + xh GEMM (+ bias fold b_ih+b_hh)
// grid.x = (B*T)/16 M-tiles; 256 threads = 8 waves; wave w does N-tiles {w, w+8}
// ---------------------------------------------------------------------------
__global__ void __launch_bounds__(256)
k_embed_gemm(const int* __restrict__ x, const float* __restrict__ emb,
             const __bf16* __restrict__ wih, const float* __restrict__ b_ih,
             const float* __restrict__ b_hh, float* __restrict__ xh) {
  __shared__ __bf16 sA[16 * EE];  // 16 gathered embedding rows, bf16 (8 KB)
  const int tid = threadIdx.x;
  const int mtile = blockIdx.x;

  // Cooperative gather+convert: each thread handles 16 contiguous elements.
  {
    const int r  = tid >> 4;          // 0..15: row within tile
    const int c0 = (tid & 15) * 16;   // 0..240: col chunk
    const int row = mtile * 16 + r;
    const float* src = emb + (long)x[row] * EE + c0;
    __builtin_prefetch(src, 0, 0);
#pragma unroll
    for (int j = 0; j < 16; j += 4) {
      v4f v = *(const v4f*)(src + j);
#pragma unroll
      for (int q = 0; q < 4; ++q) sA[r * EE + c0 + j + q] = (__bf16)v[q];
    }
  }
  __syncthreads();

  const int wave = tid >> 5;
  const int lane = tid & 31;
  const __bf16* arow = sA + (lane & 15) * EE;

  // Hoist A fragments once, reuse for both N-tiles.
  v16bf va[8];
#pragma unroll
  for (int k = 0; k < 8; ++k) va[k] = load_frag_a(arow, k, lane);

  const int mrow0 = mtile * 16 + ((lane & 16) ? 8 : 0);
#pragma unroll
  for (int i = 0; i < 2; ++i) {
    const int nt = wave + i * 8;
    const int n  = nt * 16 + (lane & 15);
    const __bf16* brow = wih + n * EE;
    v8f acc = {0.f, 0.f, 0.f, 0.f, 0.f, 0.f, 0.f, 0.f};
#pragma unroll
    for (int k = 0; k < 8; ++k)
      acc = wmma_bf16(va[k], load_frag_b(brow, k, lane), acc);
    const float bias = b_ih[n] + b_hh[n];
#pragma unroll
    for (int r = 0; r < 8; ++r)
      xh[(mrow0 + r) * HH + n] = acc[r] + bias;
  }
}

// ---------------------------------------------------------------------------
// k2: sequential RNN scan. 4 blocks, each owns 16 batch rows (one M-tile).
//   - W_hh B-fragments hoisted into registers (loop-invariant, ~128 VGPRs;
//     only 2 waves/SIMD resident so this fits easily).
//   - h double-buffered in LDS as bf16.
//   - xh[t] tile (16x256 fp32, row stride TT*HH) staged one step ahead by
//     the Tensor Data Mover into double-buffered LDS (wave 0 issues TDM,
//     s_wait_tensorcnt before the step barrier).
// ---------------------------------------------------------------------------
__global__ void __launch_bounds__(256)
k_rnn_scan(const float* __restrict__ xh, const __bf16* __restrict__ whh,
           __bf16* __restrict__ h_out) {
  __shared__ __bf16 sH[2][16 * HH];   // 2 x 8 KB
#if HAVE_TDM
  __shared__ float sX[2][16 * HH];    // 2 x 16 KB xh staging
#endif
  const int tid  = threadIdx.x;
  const int wave = tid >> 5;
  const int lane = tid & 31;
  const int mloc0 = (lane & 16) ? 8 : 0;  // local row offset for C/D layout
  const int ncol  = lane & 15;
  const int bg0   = blockIdx.x * 16;
  const float* xbase = xh + (size_t)bg0 * TT * HH;

  // Hoist all loop-invariant W_hh B fragments for this wave's two N-tiles.
  v16bf vb[2][8];
#pragma unroll
  for (int i = 0; i < 2; ++i) {
    const int n = (wave + i * 8) * 16 + ncol;
    const __bf16* brow = whh + (size_t)n * HH;
#pragma unroll
    for (int k = 0; k < 8; ++k) vb[i][k] = load_frag_b(brow, k, lane);
  }

  // h0 = 0
#pragma unroll
  for (int j = 0; j < 16; ++j) sH[0][tid * 16 + j] = (__bf16)0.f;

#if HAVE_TDM
  if (wave == 0) {
    tdm_load_xh_tile(xbase, 0, &sX[0][0]);
    __builtin_amdgcn_s_wait_tensorcnt(0);
  }
#endif
  __syncthreads();

  for (int t = 0; t < TT; ++t) {
#if HAVE_TDM
    // Kick off next timestep's xh tile while this step computes.
    if (wave == 0 && t + 1 < TT)
      tdm_load_xh_tile(xbase, t + 1, &sX[(t + 1) & 1][0]);
    const float* xt = sX[t & 1];
#endif
    const __bf16* hbuf  = sH[t & 1];
    __bf16*       hnext = sH[(t + 1) & 1];
    const __bf16* arow  = hbuf + (lane & 15) * HH;

    v8f zero = {0.f, 0.f, 0.f, 0.f, 0.f, 0.f, 0.f, 0.f};
    v8f acc[2];
    acc[0] = zero;
    acc[1] = zero;
#pragma unroll
    for (int k = 0; k < 8; ++k) {
      v16bf a = load_frag_a(arow, k, lane);   // shared A for both N-tiles
      acc[0] = wmma_bf16(a, vb[0][k], acc[0]);
      acc[1] = wmma_bf16(a, vb[1][k], acc[1]);
    }

#pragma unroll
    for (int i = 0; i < 2; ++i) {
      const int n = (wave + i * 8) * 16 + ncol;
#pragma unroll
      for (int r = 0; r < 8; ++r) {
        const int mr = mloc0 + r;
#if HAVE_TDM
        const float xv = xt[mr * HH + n];
#else
        const float xv = xh[((size_t)(bg0 + mr) * TT + t) * HH + n];
#endif
        hnext[mr * HH + n] = (__bf16)tanhf(acc[i][r] + xv);
      }
    }

#if HAVE_TDM
    if (wave == 0) __builtin_amdgcn_s_wait_tensorcnt(0);
#endif
    __syncthreads();
  }

  // TT is even: final state sits in buffer 0.
#pragma unroll
  for (int j = 0; j < 16; ++j)
    h_out[blockIdx.x * (16 * HH) + tid * 16 + j] = sH[0][tid * 16 + j];
}

// ---------------------------------------------------------------------------
// k3: MLP head. fc1 (64x128, K=256) -> relu -> fc2 (64x16, K=128). 1 block.
// ---------------------------------------------------------------------------
__global__ void __launch_bounds__(256)
k_head(const __bf16* __restrict__ hfin, const __bf16* __restrict__ w1,
       const float* __restrict__ b1, const __bf16* __restrict__ w2,
       const float* __restrict__ b2, float* __restrict__ out) {
  __shared__ __bf16 sHid[BB * 128];  // 16 KB
  const int tid = threadIdx.x;
  const int wave = tid >> 5;
  const int lane = tid & 31;
  const int mloc0 = (lane & 16) ? 8 : 0;
  const int ncol  = lane & 15;

  // fc1: 4x8 = 32 tiles of 16x16; wave handles {wave, wave+8, wave+16, wave+24}
#pragma unroll
  for (int i = 0; i < 4; ++i) {
    const int tile = wave + i * 8;
    const int mt = tile >> 3, nt = tile & 7;
    const __bf16* arow = hfin + (mt * 16 + (lane & 15)) * HH;
    const int n = nt * 16 + ncol;
    const __bf16* brow = w1 + n * HH;
    v8f acc = {0.f, 0.f, 0.f, 0.f, 0.f, 0.f, 0.f, 0.f};
#pragma unroll
    for (int k = 0; k < 8; ++k)
      acc = wmma_bf16(load_frag_a(arow, k, lane), load_frag_b(brow, k, lane), acc);
    const float bias = b1[n];
#pragma unroll
    for (int r = 0; r < 8; ++r) {
      float v = acc[r] + bias;
      v = v > 0.f ? v : 0.f;  // relu
      sHid[(mt * 16 + mloc0 + r) * 128 + n] = (__bf16)v;
    }
  }
  __syncthreads();

  // fc2: 4 tiles (mt=0..3, single N-tile since O=16), waves 0..3, K=128.
  if (wave < 4) {
    const int mt = wave;
    const __bf16* arow = sHid + (mt * 16 + (lane & 15)) * 128;
    const int n = ncol;
    const __bf16* brow = w2 + n * 128;
    v8f acc = {0.f, 0.f, 0.f, 0.f, 0.f, 0.f, 0.f, 0.f};
#pragma unroll
    for (int k = 0; k < 4; ++k)
      acc = wmma_bf16(load_frag_a(arow, k, lane), load_frag_b(brow, k, lane), acc);
    const float bias = b2[n];
#pragma unroll
    for (int r = 0; r < 8; ++r)
      out[(mt * 16 + mloc0 + r) * OO + n] = acc[r] + bias;
  }
}

// ---------------------------------------------------------------------------
// host launcher
// ---------------------------------------------------------------------------
extern "C" void kernel_launch(void* const* d_in, const int* in_sizes, int n_in,
                              void* d_out, int out_size, void* d_ws, size_t ws_size,
                              hipStream_t stream) {
  const int*   x     = (const int*)d_in[0];
  const float* emb   = (const float*)d_in[1];
  const float* W_ih  = (const float*)d_in[2];
  const float* W_hh  = (const float*)d_in[3];
  const float* b_ih  = (const float*)d_in[4];
  const float* b_hh  = (const float*)d_in[5];
  const float* fc1_w = (const float*)d_in[6];
  const float* fc1_b = (const float*)d_in[7];
  const float* fc2_w = (const float*)d_in[8];
  const float* fc2_b = (const float*)d_in[9];
  float* out = (float*)d_out;

  char* ws = (char*)d_ws;
  float* xh = (float*)ws;                                   // [B*T, H] fp32
  size_t off = (size_t)BB * TT * HH * sizeof(float);
  __bf16* whh_bf = (__bf16*)(ws + off); off += (size_t)HH * HH * 2;
  __bf16* wih_bf = (__bf16*)(ws + off); off += (size_t)HH * EE * 2;
  __bf16* w1_bf  = (__bf16*)(ws + off); off += (size_t)128 * HH * 2;
  __bf16* w2_bf  = (__bf16*)(ws + off); off += (size_t)OO * 128 * 2;
  __bf16* hfin   = (__bf16*)(ws + off);                     // [B, H] bf16

  k_f32_to_bf16<<<256, 256, 0, stream>>>(W_hh,  whh_bf, HH * HH);
  k_f32_to_bf16<<<256, 256, 0, stream>>>(W_ih,  wih_bf, HH * EE);
  k_f32_to_bf16<<<128, 256, 0, stream>>>(fc1_w, w1_bf, 128 * HH);
  k_f32_to_bf16<<<8,   256, 0, stream>>>(fc2_w, w2_bf, OO * 128);

  k_embed_gemm<<<(BB * TT) / 16, 256, 0, stream>>>(x, emb, wih_bf, b_ih, b_hh, xh);
  k_rnn_scan<<<BB / 16, 256, 0, stream>>>(xh, whh_bf, hfin);
  k_head<<<1, 256, 0, stream>>>(hfin, w1_bf, fc1_b, w2_bf, fc2_b, out);
}
